// PicoDeepLOB_32676111188551
// MI455X (gfx1250) — compile-verified, matrix operand order checked by reading, not run on hardware
//
#include <hip/hip_runtime.h>

#ifndef __has_builtin
#define __has_builtin(x) 0
#endif

#define LNUM   2
#define BSZ    2
#define SLEN   2048
#define DMODEL 768
#define HQ     12
#define HKV    4
#define HDIM   64
#define NREP   3
#define FFDIM  3072
#define MROWS  (BSZ*SLEN)   // 4096

typedef __attribute__((ext_vector_type(16))) __bf16 bf16x16;
typedef __attribute__((ext_vector_type(8)))  float  f32x8;

union BFrag { bf16x16 v; uint4 u[2]; };

__device__ __forceinline__ unsigned short f2bf(float f) {
    unsigned u = __float_as_uint(f);
    u += 0x7FFFu + ((u >> 16) & 1u);          // round-to-nearest-even
    return (unsigned short)(u >> 16);
}

__device__ __forceinline__ f32x8 vzero() {
    f32x8 z = {0.f,0.f,0.f,0.f,0.f,0.f,0.f,0.f};
    return z;
}

// ---- CDNA5 async memory->LDS path (guarded; falls back to reg staging) ----
#if __has_builtin(__builtin_amdgcn_global_load_async_to_lds_b128)
#define HAVE_ASYNC_LDS 1
typedef int v4i __attribute__((vector_size(16)));
__device__ __forceinline__ void async_b128(const void* g, void* l) {
    __builtin_amdgcn_global_load_async_to_lds_b128(
        (__attribute__((address_space(1))) v4i*)g,
        (__attribute__((address_space(3))) v4i*)l, 0, 0);
}
#else
#define HAVE_ASYNC_LDS 0
#endif

#if __has_builtin(__builtin_amdgcn_s_wait_asynccnt)
#define WAIT_ASYNC0() __builtin_amdgcn_s_wait_asynccnt(0)
#else
#define WAIT_ASYNC0() asm volatile("s_wait_asynccnt 0x0" ::: "memory")
#endif

// ---------------------------------------------------------------------------
// Tiled transpose + fp32->bf16 cast for weights: in [L][K][N] -> out [L][N][K]
// ---------------------------------------------------------------------------
__global__ __launch_bounds__(256)
void transpose_cast_k(const float* __restrict__ in, unsigned short* __restrict__ out,
                      int K, int N) {
    __shared__ float tile[32][33];
    const int nt = blockIdx.x * 32;
    const int kt = blockIdx.y * 32;
    const float* inp = in + (size_t)blockIdx.z * K * N;
    unsigned short* outp = out + (size_t)blockIdx.z * K * N;
    const int tx = threadIdx.x, ty = threadIdx.y;
    #pragma unroll
    for (int j = 0; j < 4; ++j)
        tile[ty + j*8][tx] = inp[(size_t)(kt + ty + j*8) * N + nt + tx];
    __syncthreads();
    #pragma unroll
    for (int j = 0; j < 4; ++j)
        outp[(size_t)(nt + ty + j*8) * K + kt + tx] = f2bf(tile[tx][ty + j*8]);
}

// ---------------------------------------------------------------------------
// RMSNorm over rows of length DMODEL; template selects bf16 or fp32 output
// ---------------------------------------------------------------------------
template<int BF16OUT>
__global__ __launch_bounds__(256)
void rmsnorm_k(const float* __restrict__ x, const float* __restrict__ w,
               void* __restrict__ out) {
    __shared__ float red[256];
    const int row = blockIdx.x;
    const int tid = threadIdx.x;
    const float* xr = x + (size_t)row * DMODEL;
    float s = 0.f;
    #pragma unroll
    for (int j = 0; j < 3; ++j) { float v = xr[tid + j*256]; s += v*v; }
    red[tid] = s;
    __syncthreads();
    for (int st = 128; st > 0; st >>= 1) {
        if (tid < st) red[tid] += red[tid + st];
        __syncthreads();
    }
    const float scale = rsqrtf(red[0] * (1.0f/DMODEL) + 1e-6f);
    #pragma unroll
    for (int j = 0; j < 3; ++j) {
        int c = tid + j*256;
        float v = xr[c] * scale * w[c];
        if (BF16OUT) ((unsigned short*)out)[(size_t)row*DMODEL + c] = f2bf(v);
        else         ((float*)out)[(size_t)row*DMODEL + c] = v;
    }
}

// ---------------------------------------------------------------------------
// RoPE: fp32 in [M, heads, 64] -> bf16 out
// ---------------------------------------------------------------------------
__global__ void rope_k(const float* __restrict__ in,
                       unsigned short* __restrict__ out, int heads, long npairs) {
    long i = (long)blockIdx.x * blockDim.x + threadIdx.x;
    long stride = (long)gridDim.x * blockDim.x;
    for (; i < npairs; i += stride) {
        long row = i / (heads * 32);
        int  rem = (int)(i % (heads * 32));
        int  hh  = rem >> 5;
        int  j   = rem & 31;
        int  pos = (int)(row % SLEN);
        float invf = __powf(10000.0f, -((float)(2*j)) / (float)HDIM);
        float ang = (float)pos * invf;
        float c = __cosf(ang), s = __sinf(ang);
        size_t base = ((size_t)row * heads + hh) * HDIM + 2*j;
        float t0 = in[base], t1 = in[base + 1];
        out[base]     = f2bf(t0*c - t1*s);
        out[base + 1] = f2bf(t0*s + t1*c);
    }
}

// ---------------------------------------------------------------------------
// SwiGLU gate: silu(g0)*g1 -> bf16
// ---------------------------------------------------------------------------
__global__ void silu_mul_k(const float* __restrict__ g0, const float* __restrict__ g1,
                           unsigned short* __restrict__ out, long n) {
    long i = (long)blockIdx.x * blockDim.x + threadIdx.x;
    long stride = (long)gridDim.x * blockDim.x;
    for (; i < n; i += stride) {
        float a = g0[i];
        float sl = a / (1.0f + __expf(-a));
        out[i] = f2bf(sl * g1[i]);
    }
}

// ---------------------------------------------------------------------------
// bf16 GEMM, fp32 accumulation, v_wmma_f32_16x16x32_bf16.
// A: [M][K] row-major bf16.  Bt: [N][K] row-major bf16 (pre-transposed weight).
// Block: 256 thr (8 waves), block tile 128x256, wave tile 64x64, BK=32.
// Staging: async global->LDS (double-buffered, 1 barrier/slab) when available,
// otherwise register double-buffer + global_prefetch.
// EPI: 0 = fp32, 1 = bf16, 2 = fp32 + residual, 3 = bf16 transposed ([n][m])
// ---------------------------------------------------------------------------
template<int EPI>
__global__ __launch_bounds__(256)
void gemm_bf16_wmma(const unsigned short* __restrict__ A,
                    const unsigned short* __restrict__ Bt,
                    void* __restrict__ Cout,
                    const float* __restrict__ Res,
                    int M, int N, int K) {
#if HAVE_ASYNC_LDS
    __shared__ unsigned short Ash[2][128][40];   // [m][k], 80B row stride
    __shared__ unsigned short Bsh[2][256][40];   // [n][k]
#else
    __shared__ unsigned short Ash[1][128][40];
    __shared__ unsigned short Bsh[1][256][40];
#endif

    const int tid  = threadIdx.x;
    const int wave = tid >> 5, lane = tid & 31;
    const int wm = wave >> 2, wn = wave & 3;       // 2 x 4 wave grid
    const int half = lane >> 4, l16 = lane & 15;
    const int rowBase = blockIdx.x * 128;
    const int colBase = blockIdx.y * 256;

    f32x8 acc[4][4];
    #pragma unroll
    for (int s = 0; s < 4; ++s)
        #pragma unroll
        for (int t = 0; t < 4; ++t) acc[s][t] = vzero();

    const int arow = tid >> 1, aseg = tid & 1;   // A: 2 thr/row, 16 u16 each
    const int brow = tid;                        // B: 1 thr/row, 32 u16 each

    const unsigned short* aG = A  + (size_t)(rowBase + arow) * K + aseg*16;
    const unsigned short* bG = Bt + (size_t)(colBase + brow) * K;

#if HAVE_ASYNC_LDS
    // ---- async-to-LDS double-buffered pipeline (ASYNCcnt tracked) ----
    {
        async_b128(aG,      &Ash[0][arow][aseg*16]);
        async_b128(aG + 8,  &Ash[0][arow][aseg*16 + 8]);
        #pragma unroll
        for (int j = 0; j < 4; ++j) async_b128(bG + j*8, &Bsh[0][brow][j*8]);
    }
    int it = 0;
    for (int k0 = 0; k0 < K; k0 += 32, ++it) {
        const int pb = it & 1;
        WAIT_ASYNC0();        // own async stores to LDS complete
        __syncthreads();      // all waves' loads done AND previous compute done
        if (k0 + 32 < K) {    // kick off next slab into the other buffer
            const unsigned short* aN = aG + (k0 + 32);
            const unsigned short* bN = bG + (k0 + 32);
            async_b128(aN,      &Ash[1-pb][arow][aseg*16]);
            async_b128(aN + 8,  &Ash[1-pb][arow][aseg*16 + 8]);
            #pragma unroll
            for (int j = 0; j < 4; ++j) async_b128(bN + j*8, &Bsh[1-pb][brow][j*8]);
        }
        const int kf = half * 8;
        BFrag af[4], bf[4];
        #pragma unroll
        for (int s = 0; s < 4; ++s) {
            af[s].u[0] = *(const uint4*)&Ash[pb][wm*64 + s*16 + l16][kf];
            af[s].u[1] = *(const uint4*)&Ash[pb][wm*64 + s*16 + l16][kf + 16];
        }
        #pragma unroll
        for (int t = 0; t < 4; ++t) {
            bf[t].u[0] = *(const uint4*)&Bsh[pb][wn*64 + t*16 + l16][kf];
            bf[t].u[1] = *(const uint4*)&Bsh[pb][wn*64 + t*16 + l16][kf + 16];
        }
        #pragma unroll
        for (int s = 0; s < 4; ++s)
            #pragma unroll
            for (int t = 0; t < 4; ++t)
                acc[s][t] = __builtin_amdgcn_wmma_f32_16x16x32_bf16(
                    false, af[s].v, false, bf[t].v, (short)0, acc[s][t], false, false);
    }
#else
    // ---- fallback: register double-buffer + global_prefetch ----
    uint4 aR[2], bR[4];
    aR[0] = ((const uint4*)aG)[0];
    aR[1] = ((const uint4*)aG)[1];
    #pragma unroll
    for (int j = 0; j < 4; ++j) bR[j] = ((const uint4*)bG)[j];

    for (int k0 = 0; k0 < K; k0 += 32) {
        uint4* ad = (uint4*)&Ash[0][arow][aseg*16];
        ad[0] = aR[0]; ad[1] = aR[1];
        uint4* bd = (uint4*)&Bsh[0][brow][0];
        #pragma unroll
        for (int j = 0; j < 4; ++j) bd[j] = bR[j];
        __syncthreads();

        if (k0 + 32 < K) {
            const unsigned short* aN = aG + (k0 + 32);
            const unsigned short* bN = bG + (k0 + 32);
            aR[0] = ((const uint4*)aN)[0];
            aR[1] = ((const uint4*)aN)[1];
            #pragma unroll
            for (int j = 0; j < 4; ++j) bR[j] = ((const uint4*)bN)[j];
            if (k0 + 64 < K) {
                __builtin_prefetch(aG + (k0 + 64), 0, 1);   // global_prefetch_b8
                __builtin_prefetch(bG + (k0 + 64), 0, 1);
            }
        }

        const int kf = half * 8;
        BFrag af[4], bf[4];
        #pragma unroll
        for (int s = 0; s < 4; ++s) {
            af[s].u[0] = *(const uint4*)&Ash[0][wm*64 + s*16 + l16][kf];
            af[s].u[1] = *(const uint4*)&Ash[0][wm*64 + s*16 + l16][kf + 16];
        }
        #pragma unroll
        for (int t = 0; t < 4; ++t) {
            bf[t].u[0] = *(const uint4*)&Bsh[0][wn*64 + t*16 + l16][kf];
            bf[t].u[1] = *(const uint4*)&Bsh[0][wn*64 + t*16 + l16][kf + 16];
        }
        #pragma unroll
        for (int s = 0; s < 4; ++s)
            #pragma unroll
            for (int t = 0; t < 4; ++t)
                acc[s][t] = __builtin_amdgcn_wmma_f32_16x16x32_bf16(
                    false, af[s].v, false, bf[t].v, (short)0, acc[s][t], false, false);
        __syncthreads();
    }
#endif

    #pragma unroll
    for (int s = 0; s < 4; ++s)
    #pragma unroll
    for (int t = 0; t < 4; ++t)
    #pragma unroll
    for (int e = 0; e < 8; ++e) {
        int r = rowBase + wm*64 + s*16 + half*8 + e;   // C layout: VGPR e -> row e+8*half
        int c = colBase + wn*64 + t*16 + l16;
        float v = acc[s][t][e];
        if (EPI == 2) v += Res[(size_t)r * N + c];
        if (EPI == 3)      ((unsigned short*)Cout)[(size_t)c * M + r] = f2bf(v);
        else if (EPI == 1) ((unsigned short*)Cout)[(size_t)r * N + c] = f2bf(v);
        else               ((float*)Cout)[(size_t)r * N + c] = v;
    }
}

// ---------------------------------------------------------------------------
// Causal flash attention, GQA. Block: 128 thr (4 waves), one (b, h, 64 q-rows).
// Q: bf16 [B,S,HQ,64]; K: bf16 [B,S,HKV,64]; V: bf16 TRANSPOSED [HKV*64][B*S].
// ---------------------------------------------------------------------------
__global__ __launch_bounds__(128)
void attn_fwd(const unsigned short* __restrict__ Q,
              const unsigned short* __restrict__ Kc,
              const unsigned short* __restrict__ Vt,
              unsigned short* __restrict__ O) {
    __shared__ unsigned short ksh[64][72];     // K tile row-major [kk][d]
    __shared__ unsigned short vts[64][72];     // V tile [d][kk]
    __shared__ unsigned short psh[4][16][72];  // per-wave P tile (bf16 A-frags)

    const int qblk = blockIdx.x;
    const int h    = blockIdx.y;
    const int b    = blockIdx.z;
    const int kvh  = h / NREP;
    const int tid  = threadIdx.x;
    const int wave = tid >> 5, lane = tid & 31;
    const int half = lane >> 4, l16 = lane & 15;

    // Q A-fragments (16x64 per wave), loaded once straight from global
    BFrag aq[2];
    {
        const unsigned short* qr =
            Q + (((size_t)(b*SLEN + qblk*64 + wave*16 + l16)) * HQ + h) * HDIM;
        const int kf = half * 8;
        aq[0].u[0] = *(const uint4*)(qr + kf);
        aq[0].u[1] = *(const uint4*)(qr + kf + 16);
        aq[1].u[0] = *(const uint4*)(qr + 32 + kf);
        aq[1].u[1] = *(const uint4*)(qr + 32 + kf + 16);
    }

    f32x8 oacc[4];
    #pragma unroll
    for (int t = 0; t < 4; ++t) oacc[t] = vzero();
    float mrow[8], lrow[8];
    #pragma unroll
    for (int e = 0; e < 8; ++e) { mrow[e] = -3.0e38f; lrow[e] = 0.0f; }

    const int srow = tid >> 1, sseg = tid & 1;

    for (int kb0 = 0; kb0 <= qblk; ++kb0) {
        // stage K (row-major) and V (vectorized, V pre-transposed in global)
        {
            const unsigned short* ks =
                Kc + (((size_t)(b*SLEN + kb0*64 + srow)) * HKV + kvh) * HDIM + sseg*32;
            const uint4* kp = (const uint4*)ks;
            uint4* kd = (uint4*)&ksh[srow][sseg*32];
            kd[0]=kp[0]; kd[1]=kp[1]; kd[2]=kp[2]; kd[3]=kp[3];

            const unsigned short* vs =
                Vt + (size_t)(kvh*HDIM + srow) * MROWS + (size_t)b*SLEN + kb0*64 + sseg*32;
            const uint4* vp = (const uint4*)vs;
            uint4* vd = (uint4*)&vts[srow][sseg*32];
            vd[0]=vp[0]; vd[1]=vp[1]; vd[2]=vp[2]; vd[3]=vp[3];

            if (kb0 < qblk) {  // prefetch next tile into L2
                __builtin_prefetch(ks + (size_t)64*HKV*HDIM, 0, 1);
                __builtin_prefetch(vs + 64, 0, 1);
            }
        }
        __syncthreads();

        // S = Q K^T * 1/sqrt(64): load ALL fragments first so ds loads pipeline
        f32x8 sacc[4];
        #pragma unroll
        for (int t = 0; t < 4; ++t) sacc[t] = vzero();
        {
            BFrag bkf[2][4];
            #pragma unroll
            for (int c = 0; c < 2; ++c) {
                const int kf = c*32 + half*8;
                #pragma unroll
                for (int t = 0; t < 4; ++t) {
                    bkf[c][t].u[0] = *(const uint4*)&ksh[t*16 + l16][kf];
                    bkf[c][t].u[1] = *(const uint4*)&ksh[t*16 + l16][kf + 16];
                }
            }
            #pragma unroll
            for (int c = 0; c < 2; ++c)
                #pragma unroll
                for (int t = 0; t < 4; ++t)
                    sacc[t] = __builtin_amdgcn_wmma_f32_16x16x32_bf16(
                        false, aq[c].v, false, bkf[c][t].v, (short)0, sacc[t], false, false);
        }

        // online softmax (row reductions over 16-lane groups; wave32)
        const int rowq0 = qblk*64 + wave*16 + half*8;
        #pragma unroll
        for (int e = 0; e < 8; ++e) {
            const int rq = rowq0 + e;
            float sv[4];
            #pragma unroll
            for (int t = 0; t < 4; ++t) {
                int ck = kb0*64 + t*16 + l16;
                float v = sacc[t][e] * 0.125f;
                sv[t] = (ck > rq) ? -1.0e30f : v;
            }
            float mx = fmaxf(fmaxf(sv[0], sv[1]), fmaxf(sv[2], sv[3]));
            #pragma unroll
            for (int o2 = 8; o2 >= 1; o2 >>= 1) mx = fmaxf(mx, __shfl_xor(mx, o2, 16));
            float mnew = fmaxf(mrow[e], mx);
            float corr = __expf(mrow[e] - mnew);
            float ls = 0.0f;
            #pragma unroll
            for (int t = 0; t < 4; ++t) {
                float p = __expf(sv[t] - mnew);
                ls += p;
                psh[wave][half*8 + e][t*16 + l16] = f2bf(p);
            }
            #pragma unroll
            for (int o2 = 8; o2 >= 1; o2 >>= 1) ls += __shfl_xor(ls, o2, 16);
            lrow[e] = lrow[e] * corr + ls;
            mrow[e] = mnew;
            #pragma unroll
            for (int t = 0; t < 4; ++t) oacc[t][e] *= corr;
        }
        __syncthreads();

        // O += P V: load all fragments first, then WMMA chain
        {
            BFrag ap[2], bvf[2][4];
            #pragma unroll
            for (int c = 0; c < 2; ++c) {
                const int kf = c*32 + half*8;
                ap[c].u[0] = *(const uint4*)&psh[wave][l16][kf];
                ap[c].u[1] = *(const uint4*)&psh[wave][l16][kf + 16];
                #pragma unroll
                for (int t = 0; t < 4; ++t) {
                    bvf[c][t].u[0] = *(const uint4*)&vts[t*16 + l16][kf];
                    bvf[c][t].u[1] = *(const uint4*)&vts[t*16 + l16][kf + 16];
                }
            }
            #pragma unroll
            for (int c = 0; c < 2; ++c)
                #pragma unroll
                for (int t = 0; t < 4; ++t)
                    oacc[t] = __builtin_amdgcn_wmma_f32_16x16x32_bf16(
                        false, ap[c].v, false, bvf[c][t].v, (short)0, oacc[t], false, false);
        }
        __syncthreads();
    }

    // normalize and store bf16 [B,S,H*HD]
    #pragma unroll
    for (int e = 0; e < 8; ++e) {
        float inv = 1.0f / lrow[e];
        int rq = qblk*64 + wave*16 + half*8 + e;
        #pragma unroll
        for (int t = 0; t < 4; ++t) {
            int col = h*HDIM + t*16 + l16;
            O[((size_t)(b*SLEN + rq)) * (HQ*HDIM) + col] = f2bf(oacc[t][e] * inv);
        }
    }
}

// ---------------------------------------------------------------------------
// Host driver
// ---------------------------------------------------------------------------
extern "C" void kernel_launch(void* const* d_in, const int* in_sizes, int n_in,
                              void* d_out, int out_size, void* d_ws, size_t ws_size,
                              hipStream_t stream) {
    (void)in_sizes; (void)n_in; (void)out_size; (void)ws_size;

    const float* x   = (const float*)d_in[0];
    const float* Wq  = (const float*)d_in[1];
    const float* Wk  = (const float*)d_in[2];
    const float* Wv  = (const float*)d_in[3];
    const float* Wo  = (const float*)d_in[4];
    const float* anw = (const float*)d_in[5];
    const float* w0  = (const float*)d_in[6];
    const float* w1  = (const float*)d_in[7];
    const float* w2  = (const float*)d_in[8];
    const float* snw = (const float*)d_in[9];
    const float* onw = (const float*)d_in[10];

    unsigned char* base = (unsigned char*)d_ws;
    size_t off = 0;
    auto carve = [&](size_t bytes) -> void* {
        void* p = base + off;
        off = (off + bytes + 255) & ~(size_t)255;
        return p;
    };

    const size_t M = MROWS;
    const int KVD = HKV * HDIM;  // 256
    unsigned short* wq_t = (unsigned short*)carve((size_t)LNUM*DMODEL*DMODEL*2);
    unsigned short* wk_t = (unsigned short*)carve((size_t)LNUM*DMODEL*KVD*2);
    unsigned short* wv_t = (unsigned short*)carve((size_t)LNUM*DMODEL*KVD*2);
    unsigned short* wo_t = (unsigned short*)carve((size_t)LNUM*DMODEL*DMODEL*2);
    unsigned short* w0_t = (unsigned short*)carve((size_t)LNUM*DMODEL*FFDIM*2);
    unsigned short* w1_t = (unsigned short*)carve((size_t)LNUM*DMODEL*FFDIM*2);
    unsigned short* w2_t = (unsigned short*)carve((size_t)LNUM*FFDIM*DMODEL*2);
    unsigned short* hb   = (unsigned short*)carve(M*DMODEL*2);
    float*          qf   = (float*)carve(M*DMODEL*4);
    float*          kfb  = (float*)carve(M*KVD*4);
    unsigned short* qbb  = (unsigned short*)carve(M*DMODEL*2);
    unsigned short* kbb  = (unsigned short*)carve(M*KVD*2);
    unsigned short* vtb  = (unsigned short*)carve(M*KVD*2);   // transposed [KVD][M]
    unsigned short* obb  = (unsigned short*)carve(M*DMODEL*2);
    float*          h2   = (float*)carve(M*DMODEL*4);
    unsigned short* nb   = (unsigned short*)carve(M*DMODEL*2);
    float*          g0   = (float*)carve(M*FFDIM*4);
    float*          g1   = (float*)carve(M*FFDIM*4);
    unsigned short* fb   = (unsigned short*)carve(M*FFDIM*2);
    float*          xA   = (float*)carve(M*DMODEL*4);
    float*          xB   = (float*)carve(M*DMODEL*4);

    // weight conversion + transpose: [K][N] fp32 -> [N][K] bf16
    dim3 tcb(32, 8);
    transpose_cast_k<<<dim3(DMODEL/32, DMODEL/32, LNUM), tcb, 0, stream>>>(Wq, wq_t, DMODEL, DMODEL);
    transpose_cast_k<<<dim3(KVD/32,    DMODEL/32, LNUM), tcb, 0, stream>>>(Wk, wk_t, DMODEL, KVD);
    transpose_cast_k<<<dim3(KVD/32,    DMODEL/32, LNUM), tcb, 0, stream>>>(Wv, wv_t, DMODEL, KVD);
    transpose_cast_k<<<dim3(DMODEL/32, DMODEL/32, LNUM), tcb, 0, stream>>>(Wo, wo_t, DMODEL, DMODEL);
    transpose_cast_k<<<dim3(FFDIM/32,  DMODEL/32, LNUM), tcb, 0, stream>>>(w0, w0_t, DMODEL, FFDIM);
    transpose_cast_k<<<dim3(FFDIM/32,  DMODEL/32, LNUM), tcb, 0, stream>>>(w1, w1_t, DMODEL, FFDIM);
    transpose_cast_k<<<dim3(DMODEL/32, FFDIM/32,  LNUM), tcb, 0, stream>>>(w2, w2_t, FFDIM, DMODEL);

    const float* xin = x;
    for (int i = 0; i < LNUM; ++i) {
        float* xout = (i == 0) ? xA : xB;
        const unsigned short* wqL = wq_t + (size_t)i*DMODEL*DMODEL;
        const unsigned short* wkL = wk_t + (size_t)i*DMODEL*KVD;
        const unsigned short* wvL = wv_t + (size_t)i*DMODEL*KVD;
        const unsigned short* woL = wo_t + (size_t)i*DMODEL*DMODEL;
        const unsigned short* w0L = w0_t + (size_t)i*DMODEL*FFDIM;
        const unsigned short* w1L = w1_t + (size_t)i*DMODEL*FFDIM;
        const unsigned short* w2L = w2_t + (size_t)i*FFDIM*DMODEL;

        // attn pre-norm -> bf16
        rmsnorm_k<1><<<M, 256, 0, stream>>>(xin, anw + (size_t)i*DMODEL, hb);

        // QKV projections (V written transposed for attention staging)
        gemm_bf16_wmma<0><<<dim3(M/128, DMODEL/256), 256, 0, stream>>>(hb, wqL, qf,  nullptr, M, DMODEL, DMODEL);
        gemm_bf16_wmma<0><<<dim3(M/128, KVD/256),    256, 0, stream>>>(hb, wkL, kfb, nullptr, M, KVD, DMODEL);
        gemm_bf16_wmma<3><<<dim3(M/128, KVD/256),    256, 0, stream>>>(hb, wvL, vtb, nullptr, M, KVD, DMODEL);

        // RoPE + cast to bf16
        rope_k<<<3072, 256, 0, stream>>>(qf,  qbb, HQ,  (long)M*HQ*32);
        rope_k<<<1024, 256, 0, stream>>>(kfb, kbb, HKV, (long)M*HKV*32);

        // flash attention
        attn_fwd<<<dim3(SLEN/64, HQ, BSZ), 128, 0, stream>>>(qbb, kbb, vtb, obb);

        // output projection + residual
        gemm_bf16_wmma<2><<<dim3(M/128, DMODEL/256), 256, 0, stream>>>(obb, woL, h2, xin, M, DMODEL, DMODEL);

        // ffn pre-norm -> bf16
        rmsnorm_k<1><<<M, 256, 0, stream>>>(h2, snw + (size_t)i*DMODEL, nb);

        // FFN
        gemm_bf16_wmma<0><<<dim3(M/128, FFDIM/256), 256, 0, stream>>>(nb, w0L, g0, nullptr, M, FFDIM, DMODEL);
        gemm_bf16_wmma<0><<<dim3(M/128, FFDIM/256), 256, 0, stream>>>(nb, w1L, g1, nullptr, M, FFDIM, DMODEL);
        silu_mul_k<<<4096, 256, 0, stream>>>(g0, g1, fb, (long)M*FFDIM);
        gemm_bf16_wmma<2><<<dim3(M/128, DMODEL/256), 256, 0, stream>>>(fb, w2L, xout, h2, M, DMODEL, FFDIM);

        xin = xout;
    }

    // final norm -> fp32 output
    rmsnorm_k<0><<<M, 256, 0, stream>>>(xin, onw, (float*)d_out);
}
